// DAB_25941602467880
// MI455X (gfx1250) — matrix operand order repaired and checked by reading, not compiled, not run on hardware
//
#include <hip/hip_runtime.h>

// Deformable-attention block for MI455X (gfx1250, wave32).
// GEMM parts run on v_wmma_f32_16x16x32_bf16 (bf16 in, f32 accumulate):
// fp32 matrix rate would make the 22 GFLOP compute-bound (~70us) while bf16
// WMMA brings it to the ~11us HBM floor (23.3 TB/s). im2col patches are staged
// in LDS in B-fragment-friendly layout; weights are pre-converted to bf16
// row-major so A fragments are two contiguous b128 loads per K-step.

#define BB    8
#define C_    64
#define HH    128
#define WW    128
#define KT    9          // 3x3 taps
#define KDIM  576        // C_*KT, == 18*32 exactly (no K remainder)
#define NSTEP 18         // KDIM/32
#define HP    130
#define WP    130
#define COFF  18         // offset-conv channels
#define MOFF  32         // padded M for offset conv
#define CHUNK 32         // pixels per block along W
#define PSTR  592        // LDS patch stride in bf16 elems (1184B, 32B aligned)

typedef __attribute__((ext_vector_type(16))) __bf16 v16bf;
typedef __attribute__((ext_vector_type(8)))  __bf16 v8bf;
typedef __attribute__((ext_vector_type(8)))  float  v8f;

static __device__ __forceinline__ __bf16 f2bf(float f) {
  unsigned u = __builtin_bit_cast(unsigned, f);
  u += 0x7FFFu + ((u >> 16) & 1u);                 // round-to-nearest-even
  unsigned short hs = (unsigned short)(u >> 16);
  return __builtin_bit_cast(__bf16, hs);
}

// ---- shared WMMA macro-tile: one 16(M)x16(N) f32 tile, K = 576 -------------
// A lane layout (16-bit A 16x32): M = lane&15, K-half = lane>>4;
//   elems 0..7  = Wrow[k0 + 8h .. +7], elems 8..15 = Wrow[k0+16+8h .. +7]
// B lane layout (16-bit B 32x16): N = lane&15; same K order as A
static __device__ __forceinline__ v8f conv_tile(const __bf16* __restrict__ Wrow,
                                                const __bf16* pB, int h) {
  v8f acc = {};
  __builtin_prefetch(Wrow, 0, 0);                  // global_prefetch_b8
#pragma unroll
  for (int ks = 0; ks < NSTEP; ++ks) {
    const int k0 = ks * 32;
    v8bf a0 = *(const v8bf*)(Wrow + k0 + 8 * h);
    v8bf a1 = *(const v8bf*)(Wrow + k0 + 16 + 8 * h);
    v16bf A = __builtin_shufflevector(a0, a1, 0, 1, 2, 3, 4, 5, 6, 7,
                                      8, 9, 10, 11, 12, 13, 14, 15);
    v8bf b0 = *(const v8bf*)(pB + k0 + 8 * h);       // 16B aligned LDS loads
    v8bf b1 = *(const v8bf*)(pB + k0 + 16 + 8 * h);
    v16bf Bv = __builtin_shufflevector(b0, b1, 0, 1, 2, 3, 4, 5, 6, 7,
                                       8, 9, 10, 11, 12, 13, 14, 15);
    acc = __builtin_amdgcn_wmma_f32_16x16x32_bf16(
        false, A, false, Bv, (short)0, acc, false, false);
  }
  return acc;
}

// ---- kernel 0a: zero-pad x -> x_pad [B,64,130,130] -------------------------
__global__ void k_pad(const float* __restrict__ x, float* __restrict__ xp,
                      int total) {
  int i = blockIdx.x * blockDim.x + threadIdx.x;
  if (i >= total) return;
  int wp = i % WP;
  int t = i / WP;
  int hp = t % HP;
  t /= HP;                                          // t = b*C_ + c
  float v = 0.f;
  if (hp >= 1 && hp <= HH && wp >= 1 && wp <= WW)
    v = x[(t * HH + (hp - 1)) * WW + (wp - 1)];
  xp[i] = v;
}

// ---- kernel 0b: bf16 row-major weight matrices -----------------------------
// Wo: offset conv, M padded 18->32 (zero rows). Wc: attention conv 64x576.
// Wd: grouped deform weights zero-expanded to dense block-diagonal 64x576.
__global__ void k_weights(const float* __restrict__ ow,
                          const float* __restrict__ cw,
                          const float* __restrict__ dw,
                          __bf16* __restrict__ Wo, __bf16* __restrict__ Wc,
                          __bf16* __restrict__ Wd) {
  int i = blockIdx.x * blockDim.x + threadIdx.x;
  const int NO = MOFF * KDIM, NC = C_ * KDIM;
  if (i < NO) {
    int m = i / KDIM, k = i - m * KDIM;
    Wo[i] = f2bf(m < COFF ? ow[m * KDIM + k] : 0.f);
  } else if (i < NO + NC) {
    int j = i - NO;
    Wc[j] = f2bf(cw[j]);
  } else if (i < NO + 2 * NC) {
    int j = i - NO - NC;
    int co = j / KDIM, k = j - co * KDIM;
    int ci = k / KT, tap = k - ci * KT;
    float v = ((ci >> 3) == (co >> 3)) ? dw[(co * 8 + (ci & 7)) * KT + tap] : 0.f;
    Wd[j] = f2bf(v);
  }
}

// ---- kernel 1: offset conv (64->18) + bias + PReLU ------------------------
__global__ void k_offconv(const float* __restrict__ xp,
                          const __bf16* __restrict__ Wo,
                          const float* __restrict__ ob,
                          const float* __restrict__ pa,
                          float* __restrict__ off) {
  __shared__ alignas(32) __bf16 patch[CHUNK * PSTR];
  const int b = blockIdx.z, hrow = blockIdx.y, w0 = blockIdx.x * CHUNK;
  const int tid = threadIdx.x;
  for (int i = tid; i < CHUNK * KDIM; i += blockDim.x) {
    int n = i & (CHUNK - 1), t = i >> 5;
    int ci = t / KT, tap = t - ci * KT, ky = tap / 3, kx = tap - ky * 3;
    float v = xp[((b * C_ + ci) * HP + hrow + ky) * WP + w0 + n + kx];
    patch[n * PSTR + t] = f2bf(v);
  }
  __syncthreads();
  const int wave = tid >> 5, lane = tid & 31;
  const int mt = wave & 1, nt = wave >> 1;           // 4 waves: 2 M x 2 N tiles
  const int hh = lane >> 4, mn = lane & 15;
  v8f acc = conv_tile(Wo + (mt * 16 + mn) * KDIM, patch + (nt * 16 + mn) * PSTR, hh);
  const float a0 = pa[0];
  const int pix = w0 + nt * 16 + mn;
#pragma unroll
  for (int r = 0; r < 8; ++r) {
    int co = mt * 16 + r + 8 * hh;
    if (co < COFF) {
      float v = acc[r] + ob[co];
      v = v >= 0.f ? v : a0 * v;
      off[((b * COFF + co) * HH + hrow) * WW + pix] = v;
    }
  }
}

// ---- kernel 2a: attention conv (64->64), write logits ---------------------
__global__ void k_attconv(const float* __restrict__ xp,
                          const __bf16* __restrict__ Wc,
                          float* __restrict__ att) {
  __shared__ alignas(32) __bf16 patch[CHUNK * PSTR];
  const int b = blockIdx.z, hrow = blockIdx.y, w0 = blockIdx.x * CHUNK;
  const int tid = threadIdx.x;
  for (int i = tid; i < CHUNK * KDIM; i += blockDim.x) {
    int n = i & (CHUNK - 1), t = i >> 5;
    int ci = t / KT, tap = t - ci * KT, ky = tap / 3, kx = tap - ky * 3;
    patch[n * PSTR + t] =
        f2bf(xp[((b * C_ + ci) * HP + hrow + ky) * WP + w0 + n + kx]);
  }
  __syncthreads();
  const int wave = tid >> 5, lane = tid & 31;
  const int mt = wave & 3, nt = wave >> 2;           // 8 waves: 4 M x 2 N tiles
  const int hh = lane >> 4, mn = lane & 15;
  v8f acc = conv_tile(Wc + (mt * 16 + mn) * KDIM, patch + (nt * 16 + mn) * PSTR, hh);
  const int pix = w0 + nt * 16 + mn;
#pragma unroll
  for (int r = 0; r < 8; ++r) {
    int co = mt * 16 + r + 8 * hh;
    att[((b * C_ + co) * HH + hrow) * WW + pix] = acc[r];
  }
}

// ---- kernel 2b: in-place softmax over W (one wave32 per row) --------------
__global__ void k_softmax(float* __restrict__ att) {
  const int row = blockIdx.x;                        // B*C_*H rows of 128
  const int lane = threadIdx.x;
  float4 v = ((const float4*)(att + row * WW))[lane];
  float m = fmaxf(fmaxf(v.x, v.y), fmaxf(v.z, v.w));
#pragma unroll
  for (int d = 16; d; d >>= 1) m = fmaxf(m, __shfl_xor(m, d, 32));
  float ex = __expf(v.x - m), ey = __expf(v.y - m);
  float ez = __expf(v.z - m), ew = __expf(v.w - m);
  float s = ex + ey + ez + ew;
#pragma unroll
  for (int d = 16; d; d >>= 1) s += __shfl_xor(s, d, 32);
  float inv = 1.f / s;
  float4 o = {ex * inv, ey * inv, ez * inv, ew * inv};
  ((float4*)(att + row * WW))[lane] = o;
}

// ---- kernel 3: deform conv (bilinear gather -> WMMA) + atten*feat + x -----
__global__ void k_deform(const float* __restrict__ x,
                         const float* __restrict__ off,
                         const __bf16* __restrict__ Wd,
                         const float* __restrict__ db,
                         const float* __restrict__ att,
                         float* __restrict__ out) {
  __shared__ alignas(32) __bf16 patch[CHUNK * PSTR];
  __shared__ float ysl[KT * CHUNK], xsl[KT * CHUNK];
  const int b = blockIdx.z, hrow = blockIdx.y, w0 = blockIdx.x * CHUNK;
  const int tid = threadIdx.x;
  if (tid < KT * CHUNK) {                            // sample coords per (tap,n)
    int tap = tid >> 5, n = tid & 31;
    float dy = off[((b * COFF + 2 * tap) * HH + hrow) * WW + w0 + n];
    float dx = off[((b * COFF + 2 * tap + 1) * HH + hrow) * WW + w0 + n];
    ysl[tid] = (float)(hrow - 1 + tap / 3) + dy;
    xsl[tid] = (float)(w0 + n - 1 + (tap % 3)) + dx;
  }
  __syncthreads();
  for (int i = tid; i < CHUNK * KDIM; i += blockDim.x) {
    int n = i & (CHUNK - 1), t = i >> 5;
    int ci = t / KT, tap = t - ci * KT;
    float yy = ysl[tap * CHUNK + n], xx = xsl[tap * CHUNK + n];
    float y0 = floorf(yy), x0 = floorf(xx);
    float dy = yy - y0, dx = xx - x0;
    int iy = (int)y0, ix = (int)x0;
    const float* pl = x + (b * C_ + ci) * (HH * WW);
    float v = 0.f;
    float w00 = (1.f - dy) * (1.f - dx), w01 = (1.f - dy) * dx;
    float w10 = dy * (1.f - dx), w11 = dy * dx;
    if ((unsigned)iy < HH) {
      if ((unsigned)ix < WW)       v += w00 * pl[iy * WW + ix];
      if ((unsigned)(ix + 1) < WW) v += w01 * pl[iy * WW + ix + 1];
    }
    if ((unsigned)(iy + 1) < HH) {
      if ((unsigned)ix < WW)       v += w10 * pl[(iy + 1) * WW + ix];
      if ((unsigned)(ix + 1) < WW) v += w11 * pl[(iy + 1) * WW + ix + 1];
    }
    patch[n * PSTR + t] = f2bf(v);
  }
  __syncthreads();
  const int wave = tid >> 5, lane = tid & 31;
  const int mt = wave & 3, nt = wave >> 2;
  const int hh = lane >> 4, mn = lane & 15;
  v8f acc = conv_tile(Wd + (mt * 16 + mn) * KDIM, patch + (nt * 16 + mn) * PSTR, hh);
  const int pix = w0 + nt * 16 + mn;
#pragma unroll
  for (int r = 0; r < 8; ++r) {
    int co = mt * 16 + r + 8 * hh;
    int idx = ((b * C_ + co) * HH + hrow) * WW + pix;
    float feat = acc[r] + db[co];
    out[idx] = att[idx] * feat + x[idx];
  }
}

extern "C" void kernel_launch(void* const* d_in, const int* in_sizes, int n_in,
                              void* d_out, int out_size, void* d_ws,
                              size_t ws_size, hipStream_t stream) {
  (void)in_sizes; (void)n_in; (void)out_size; (void)ws_size;
  const float* x  = (const float*)d_in[0];
  const float* ow = (const float*)d_in[1];  // [18,64,3,3] -> row-major 18x576
  const float* ob = (const float*)d_in[2];
  const float* pa = (const float*)d_in[3];
  const float* dw = (const float*)d_in[4];  // [64,8,3,3]
  const float* db = (const float*)d_in[5];
  const float* cw = (const float*)d_in[6];  // [64,64,3,3] -> 64x576
  float* out = (float*)d_out;

  // workspace carve (~78 MB total)
  char* ws = (char*)d_ws;
  size_t o = 0;
  auto carve = [&](size_t bytes) {
    char* p = ws + o;
    o = (o + bytes + 255) & ~(size_t)255;
    return p;
  };
  float*  xp  = (float*) carve(sizeof(float)  * BB * C_ * HP * WP);
  float*  off = (float*) carve(sizeof(float)  * BB * COFF * HH * WW);
  float*  att = (float*) carve(sizeof(float)  * BB * C_ * HH * WW);
  __bf16* Wo  = (__bf16*)carve(sizeof(__bf16) * MOFF * KDIM);
  __bf16* Wc  = (__bf16*)carve(sizeof(__bf16) * C_ * KDIM);
  __bf16* Wd  = (__bf16*)carve(sizeof(__bf16) * C_ * KDIM);

  const int totalPad = BB * C_ * HP * WP;
  k_pad<<<(totalPad + 255) / 256, 256, 0, stream>>>(x, xp, totalPad);
  const int totalW = MOFF * KDIM + 2 * C_ * KDIM;
  k_weights<<<(totalW + 255) / 256, 256, 0, stream>>>(ow, cw, dw, Wo, Wc, Wd);

  dim3 grid(WW / CHUNK, HH, BB);                     // (4,128,8)
  k_offconv<<<grid, 128, 0, stream>>>(xp, Wo, ob, pa, off);
  k_attconv<<<grid, 256, 0, stream>>>(xp, Wc, att);
  k_softmax<<<BB * C_ * HH, 32, 0, stream>>>(att);
  k_deform<<<grid, 256, 0, stream>>>(x, off, Wd, db, att, out);
}
